// KANtoMLP_49950469653194
// MI455X (gfx1250) — matrix operand order repaired and checked by reading, not compiled, not run on hardware
//
#include <hip/hip_runtime.h>
#include <hip/hip_bf16.h>

typedef __attribute__((ext_vector_type(16))) __bf16 v16bf;
typedef __attribute__((ext_vector_type(8)))  float  v8f;
typedef __attribute__((ext_vector_type(4)))  int    v4i;

#define BM 64            // block tile M
#define BN 128           // block tile N
#define KC 64            // K chunk per LDS stage (2 WMMA K-steps)

// Async VMEM->LDS path (gfx1250 GLOBAL_LOAD_ASYNC_TO_LDS, ASYNCcnt-tracked)
#if defined(__has_builtin)
# if __has_builtin(__builtin_amdgcn_global_load_async_to_lds_b128) && \
     __has_builtin(__builtin_amdgcn_s_wait_asynccnt)
#  define KAN_ASYNC 1
# endif
#endif
#ifndef KAN_ASYNC
# define KAN_ASYNC 0
#endif

#if KAN_ASYNC
typedef __attribute__((address_space(1))) v4i gas_v4i;   // global int4
typedef __attribute__((address_space(3))) v4i las_v4i;   // LDS int4

__device__ __forceinline__ void async_copy16(const __bf16* g, __bf16* l) {
    __builtin_amdgcn_global_load_async_to_lds_b128(
        (gas_v4i*)g, (las_v4i*)l, /*offset=*/0, /*cpol=*/0);
}
#endif

// ---------------------------------------------------------------------------
// KAN feature expansion: out9[0] = silu(x), out9[1..8] = cubic B-spline bases
// grid: 12 knots, h = 0.4 over [-1,1], spline order 3, 8 coefficients.
// ---------------------------------------------------------------------------
__device__ __forceinline__ void kan_features(float x, float* out9) {
    const float h = 0.4f;
    float g[12];
#pragma unroll
    for (int i = 0; i < 12; ++i) g[i] = -1.0f + (float)(i - 3) * h;

    float b[11];
#pragma unroll
    for (int j = 0; j < 11; ++j) b[j] = (x >= g[j] && x < g[j + 1]) ? 1.0f : 0.0f;

#pragma unroll
    for (int p = 1; p <= 3; ++p) {
#pragma unroll
        for (int j = 0; j < 11; ++j) {
            if (j < 11 - p) {
                float left  = (x - g[j]) / (g[j + p] - g[j]);
                float right = (g[j + p + 1] - x) / (g[j + p + 1] - g[j + 1]);
                b[j] = left * b[j] + right * b[j + 1];
            }
        }
    }
    out9[0] = x / (1.0f + __expf(-x));   // SiLU base path
#pragma unroll
    for (int c = 0; c < 8; ++c) out9[1 + c] = b[c];
}

// ---------------------------------------------------------------------------
// Pack weights: wp[o][f*9+0] = base_w[o][f]; wp[o][f*9+1+c] = spline_w[o][f][c]*scaler
// ---------------------------------------------------------------------------
__global__ void kan_pack_w(const float* __restrict__ base_w,
                           const float* __restrict__ spline_w,
                           const float* __restrict__ scaler,
                           __bf16* __restrict__ wp, long total) {
    long idx = (long)blockIdx.x * blockDim.x + threadIdx.x;   // over OUT*IN
    if (idx >= total) return;
    float s = scaler[idx];
    __bf16* dst = wp + idx * 9;
    dst[0] = (__bf16)base_w[idx];
    const float* sw = spline_w + idx * 8;
#pragma unroll
    for (int c = 0; c < 8; ++c) dst[1 + c] = (__bf16)(sw[c] * s);
}

// ---------------------------------------------------------------------------
// Expand activations once: X[M][Kin] fp32 -> A[M][Kin*9] bf16 (silu + 8 bases).
// ---------------------------------------------------------------------------
__global__ void kan_expand(const float* __restrict__ X,
                           __bf16* __restrict__ A, long total) {
    long idx = (long)blockIdx.x * blockDim.x + threadIdx.x;   // over M*Kin
    if (idx >= total) return;
    float feat[9];
    kan_features(X[idx], feat);
    __bf16* dst = A + idx * 9;
#pragma unroll
    for (int c = 0; c < 9; ++c) dst[c] = (__bf16)feat[c];
}

// ---------------------------------------------------------------------------
// bf16 GEMM: Y[M][N] = A[M][K] * W[N][K]^T, f32 accumulate, optional GELU.
// 256 threads = 8 waves (2x4). Each wave: 32x32 output = 2x2 WMMA fragments.
// Double-buffered LDS K-chunks of 64, filled by async VMEM->LDS (ASYNCcnt)
// or a register-staged pipeline when the async builtins are unavailable.
// ---------------------------------------------------------------------------
__global__ __launch_bounds__(256)
void kan_gemm_wmma(const __bf16* __restrict__ A,
                   const __bf16* __restrict__ W,
                   float* __restrict__ Y,
                   int M, int N, int K, int gelu) {
    __shared__ __bf16 s_a[2][BM * KC];   // 2 * 8 KB
    __shared__ __bf16 s_b[2][BN * KC];   // 2 * 16 KB

    const int tid  = threadIdx.x;
    const int wave = tid >> 5;           // 0..7
    const int lane = tid & 31;

    const int mBase = blockIdx.y * BM;
    const int nBase = blockIdx.x * BN;

    const int wm = (wave >> 2) * 32;     // wave row offset (0 or 32)
    const int wn = (wave & 3) * 32;      // wave col offset (0..96)

    const int g  = lane >> 4;            // half-wave group (0/1)
    const int lr = lane & 15;            // row/col within fragment

    v8f acc00 = {}, acc01 = {}, acc10 = {}, acc11 = {};

    const int nch = K / KC;

    // per-thread copy slots: 2 A chunks + 4 B chunks of 16B each
    int arow[2], akc[2], brow[4], bkc[4];
#pragma unroll
    for (int it = 0; it < 2; ++it) {
        int c = tid + it * 256;
        arow[it] = c >> 3;          // / (KC/8)
        akc[it]  = (c & 7) * 8;
    }
#pragma unroll
    for (int it = 0; it < 4; ++it) {
        int c = tid + it * 256;
        brow[it] = c >> 3;
        bkc[it]  = (c & 7) * 8;
    }

#if KAN_ASYNC
    // ---- async pipeline: issue chunk into LDS, ASYNCcnt-tracked -----------
    auto issue = [&](int ch, int buf) {
        const long kc0 = (long)ch * KC;
#pragma unroll
        for (int it = 0; it < 2; ++it)
            async_copy16(A + (long)(mBase + arow[it]) * K + kc0 + akc[it],
                         &s_a[buf][arow[it] * KC + akc[it]]);
#pragma unroll
        for (int it = 0; it < 4; ++it)
            async_copy16(W + (long)(nBase + brow[it]) * K + kc0 + bkc[it],
                         &s_b[buf][brow[it] * KC + bkc[it]]);
    };
    issue(0, 0);
#else
    // ---- fallback: register-staged pipeline -------------------------------
    uint4 ra[2], rb[4];
    auto gload = [&](int ch) {
        const long kc0 = (long)ch * KC;
#pragma unroll
        for (int it = 0; it < 2; ++it)
            ra[it] = *(const uint4*)(A + (long)(mBase + arow[it]) * K + kc0 + akc[it]);
#pragma unroll
        for (int it = 0; it < 4; ++it)
            rb[it] = *(const uint4*)(W + (long)(nBase + brow[it]) * K + kc0 + bkc[it]);
    };
    auto lstore = [&](int buf) {
#pragma unroll
        for (int it = 0; it < 2; ++it)
            *(uint4*)(&s_a[buf][arow[it] * KC + akc[it]]) = ra[it];
#pragma unroll
        for (int it = 0; it < 4; ++it)
            *(uint4*)(&s_b[buf][brow[it] * KC + bkc[it]]) = rb[it];
    };
    gload(0);
#endif

    for (int ch = 0; ch < nch; ++ch) {
        const int cur = ch & 1;

#if KAN_ASYNC
        if (ch + 1 < nch) {
            issue(ch + 1, cur ^ 1);                 // 6 new asyncs in flight
            __builtin_amdgcn_s_wait_asynccnt(6);    // chunk `ch` landed (in-order)
        } else {
            __builtin_amdgcn_s_wait_asynccnt(0);
        }
        __syncthreads();
#else
        lstore(cur);                                 // drain regs of chunk `ch`
        if (ch + 1 < nch) gload(ch + 1);             // overlap with WMMA below
        __syncthreads();
#endif

        // ---- 2 K-steps x (2x2 fragments) = 8 WMMA per chunk ---------------
#pragma unroll
        for (int kk = 0; kk < KC; kk += 32) {
            union { v16bf v; uint4 q[2]; } a0, a1, b0, b1;
            const __bf16* ap0 = &s_a[cur][(wm + lr)      * KC + kk + g * 8];
            const __bf16* ap1 = &s_a[cur][(wm + 16 + lr) * KC + kk + g * 8];
            a0.q[0] = *(const uint4*)(ap0);
            a0.q[1] = *(const uint4*)(ap0 + 16);
            a1.q[0] = *(const uint4*)(ap1);
            a1.q[1] = *(const uint4*)(ap1 + 16);
            const __bf16* bp0 = &s_b[cur][(wn + lr)      * KC + kk + g * 8];
            const __bf16* bp1 = &s_b[cur][(wn + 16 + lr) * KC + kk + g * 8];
            b0.q[0] = *(const uint4*)(bp0);
            b0.q[1] = *(const uint4*)(bp0 + 16);
            b1.q[0] = *(const uint4*)(bp1);
            b1.q[1] = *(const uint4*)(bp1 + 16);

            acc00 = __builtin_amdgcn_wmma_f32_16x16x32_bf16(
                        false, a0.v, false, b0.v, (short)0, acc00, false, false);
            acc01 = __builtin_amdgcn_wmma_f32_16x16x32_bf16(
                        false, a0.v, false, b1.v, (short)0, acc01, false, false);
            acc10 = __builtin_amdgcn_wmma_f32_16x16x32_bf16(
                        false, a1.v, false, b0.v, (short)0, acc10, false, false);
            acc11 = __builtin_amdgcn_wmma_f32_16x16x32_bf16(
                        false, a1.v, false, b1.v, (short)0, acc11, false, false);
        }
        __syncthreads();
    }

    // ---- store 4 accumulators (fragment M = r + 8*g, N = lr) --------------
    v8f* accs[4] = { &acc00, &acc01, &acc10, &acc11 };
#pragma unroll
    for (int i = 0; i < 2; ++i) {
#pragma unroll
        for (int j = 0; j < 2; ++j) {
            v8f a = *accs[i * 2 + j];
#pragma unroll
            for (int r = 0; r < 8; ++r) {
                int row = mBase + wm + 16 * i + 8 * g + r;
                int col = nBase + wn + 16 * j + lr;
                float v = a[r];
                if (gelu) v = 0.5f * v * (1.0f + erff(v * 0.70710678118654752f));
                Y[(long)row * N + col] = v;
            }
        }
    }
}

// ---------------------------------------------------------------------------
extern "C" void kernel_launch(void* const* d_in, const int* in_sizes, int n_in,
                              void* d_out, int out_size, void* d_ws, size_t ws_size,
                              hipStream_t stream) {
    const float* x            = (const float*)d_in[0];  // [4,1024,768]
    const float* fc_base_w    = (const float*)d_in[1];  // [3072,768]
    const float* fc_spline_w  = (const float*)d_in[2];  // [3072,768,8]
    const float* fc_scaler    = (const float*)d_in[3];  // [3072,768]
    const float* pj_base_w    = (const float*)d_in[4];  // [768,3072]
    const float* pj_spline_w  = (const float*)d_in[5];  // [768,3072,8]
    const float* pj_scaler    = (const float*)d_in[6];  // [768,3072]
    float* out = (float*)d_out;                          // [4,1024,768]

    const int Ntok = 4096, H = 768, F = 3072;
    const long K1 = (long)H * 9;   // 6912
    const long K2 = (long)F * 9;   // 27648

    // workspace carve-up (256-byte aligned)
    char* ws = (char*)d_ws;
    size_t off = 0;
    auto carve = [&](size_t bytes) {
        void* p = ws + off;
        off = (off + bytes + 255) & ~(size_t)255;
        return p;
    };
    __bf16* W1p  = (__bf16*)carve((size_t)F * H * 9 * sizeof(__bf16));  // 42.5 MB
    __bf16* W2p  = (__bf16*)carve((size_t)H * F * 9 * sizeof(__bf16));  // 42.5 MB
    float*  H1   = (float*)carve((size_t)Ntok * F * sizeof(float));     // 50.3 MB
    __bf16* Abuf = (__bf16*)carve((size_t)Ntok * K2 * sizeof(__bf16));  // 226.5 MB
    // Abuf serves as A1 (Ntok x K1) for layer 1, then is overwritten as A2.

    // pack weights (fold scaler, interleave base+spline, cast to bf16)
    {
        long e = (long)F * H;
        kan_pack_w<<<dim3((e + 255) / 256), dim3(256), 0, stream>>>(
            fc_base_w, fc_spline_w, fc_scaler, W1p, e);
        kan_pack_w<<<dim3((e + 255) / 256), dim3(256), 0, stream>>>(
            pj_base_w, pj_spline_w, pj_scaler, W2p, e);
    }

    // layer 1: expand once, then pure bf16 WMMA GEMM with fused exact GELU
    {
        long e = (long)Ntok * H;
        kan_expand<<<dim3((e + 255) / 256), dim3(256), 0, stream>>>(x, Abuf, e);
        kan_gemm_wmma<<<dim3(F / BN, Ntok / BM), dim3(256), 0, stream>>>(
            Abuf, W1p, H1, Ntok, F, (int)K1, /*gelu=*/1);
    }

    // layer 2: expand GELU output, GEMM to final projection
    {
        long e = (long)Ntok * F;
        kan_expand<<<dim3((e + 255) / 256), dim3(256), 0, stream>>>(H1, Abuf, e);
        kan_gemm_wmma<<<dim3(H / BN, Ntok / BM), dim3(256), 0, stream>>>(
            Abuf, W2p, out, Ntok, H, (int)K2, /*gelu=*/0);
    }
}
